// TemplMatch_87351044866737
// MI455X (gfx1250) — matrix operand order repaired and checked by reading, not compiled, not run on hardware
//
#include <hip/hip_runtime.h>
#include <hip/hip_bf16.h>
#include <math.h>
#include <stdint.h>

typedef __attribute__((ext_vector_type(16))) _Float16 v16h;
typedef __attribute__((ext_vector_type(8)))  _Float16 v8h;
typedef __attribute__((ext_vector_type(4)))  _Float16 v4h;
typedef __attribute__((ext_vector_type(8)))  float    v8f;
typedef unsigned int u32x4 __attribute__((ext_vector_type(4)));
typedef int          i32x4 __attribute__((ext_vector_type(4)));
typedef int          i32x8 __attribute__((ext_vector_type(8)));

#define BATCH 32
#define HIN   512
#define WIN   512
#define TDIM  64
#define HOUT  449
#define TN    4096
#define EPSF  1e-8f

// workspace layout (offsets 256B aligned except OFF_XH (128B))
#define OFF_TVAR 0
#define OFF_B    256
#define SZ_B     (64*3*32*16*2)                         // 196608 B Toeplitz B frags
#define OFF_H    (OFF_B + SZ_B)                         // 196864
#define SZ_H     ((size_t)BATCH*HIN*HOUT*sizeof(float)) // 29425664 B
#define OFF_H2   (OFF_H + SZ_H)
#define OFF_D    (OFF_H2 + SZ_H)
#define SZ_D     ((size_t)BATCH*HOUT*HOUT*sizeof(float))// 25804928 B
#define OFF_XH   (OFF_D + SZ_D)                         // f16 copy of X: 16777216 B

// ---------------------------------------------------------------------------
// Kernel 1: template mean/var + Toeplitz WMMA-B fragments.
// Layout [dy][c][lane][16 halves]; lane l: n=l&15 (N), h=l>>4, elem e: K=16h+e,
// value = Tz[dy][32c + K - n] (0 outside [0,64)). (B: K x N, sparse-B layout)
// ---------------------------------------------------------------------------
__global__ void tm_prep(const float* __restrict__ tpl,
                        float* __restrict__ wsTvar,
                        _Float16* __restrict__ wsB) {
    __shared__ float tz[TN];
    __shared__ float red[256];
    const int tid = threadIdx.x;

    float s = 0.f;
    for (int i = tid; i < TN; i += 256) { float v = tpl[i]; tz[i] = v; s += v; }
    red[tid] = s; __syncthreads();
    for (int off = 128; off > 0; off >>= 1) {
        if (tid < off) red[tid] += red[tid + off];
        __syncthreads();
    }
    const float mean = red[0] * (1.f / (float)TN);
    __syncthreads();

    float s2 = 0.f;
    for (int i = tid; i < TN; i += 256) { float v = tz[i] - mean; tz[i] = v; s2 += v * v; }
    __syncthreads();
    red[tid] = s2; __syncthreads();
    for (int off = 128; off > 0; off >>= 1) {
        if (tid < off) red[tid] += red[tid + off];
        __syncthreads();
    }
    if (tid == 0) wsTvar[0] = red[0] + EPSF;
    __syncthreads();

    for (int idx = tid; idx < 64 * 3 * 32; idx += 256) {   // idx = dy*96 + c*32 + lane
        const int lane = idx & 31;
        const int c    = (idx >> 5) % 3;
        const int dy   = idx / 96;
        const int n = lane & 15, h = lane >> 4;
        _Float16* dst = wsB + (size_t)idx * 16;
        #pragma unroll
        for (int e = 0; e < 16; ++e) {
            const int t = 32 * c + 16 * h + e - n;
            const float v = (t >= 0 && t < TDIM) ? tz[dy * TDIM + t] : 0.f;
            dst[e] = (_Float16)v;
        }
    }
}

// ---------------------------------------------------------------------------
// Kernel 2: X (fp32) -> Xh (f16), once; Xh is L2-resident (16.8 MB << 192 MB)
// ---------------------------------------------------------------------------
__global__ void tm_cvt(const float* __restrict__ X, _Float16* __restrict__ Xh) {
    const size_t i = ((size_t)blockIdx.x * blockDim.x + threadIdx.x) * 4;
    if (i >= (size_t)BATCH * HIN * WIN) return;
    const float4 v = *(const float4*)(X + i);
    v4h o = { (_Float16)v.x, (_Float16)v.y, (_Float16)v.z, (_Float16)v.w };
    *(v4h*)(Xh + i) = o;
}

// ---------------------------------------------------------------------------
// Kernel 3: horizontal 64-wide box sums of X and X^2 (one block per image row)
// ---------------------------------------------------------------------------
__global__ void tm_hbox(const float* __restrict__ X,
                        float* __restrict__ H, float* __restrict__ H2) {
    __shared__ float row[WIN], row2[WIN];
    const int bid = blockIdx.x;
    const int b = bid >> 9, y = bid & 511;
    const int tid = threadIdx.x;
    const float* xp = X + ((size_t)b * HIN + y) * WIN;
    for (int i = tid; i < WIN; i += 128) { float v = xp[i]; row[i] = v; row2[i] = v * v; }
    __syncthreads();
    const size_t ob = ((size_t)b * HIN + y) * HOUT;
    for (int x = tid; x < HOUT; x += 128) {
        float s = 0.f, s2 = 0.f;
        for (int d = 0; d < TDIM; ++d) { s += row[x + d]; s2 += row2[x + d]; }
        H[ob + x] = s; H2[ob + x] = s2;
    }
}

// ---------------------------------------------------------------------------
// Kernel 4: vertical sliding 64-box sum -> dinv = 1/sqrt(tvar * img_var)
// ---------------------------------------------------------------------------
__global__ void tm_vbox(const float* __restrict__ H, const float* __restrict__ H2,
                        const float* __restrict__ wsTvar, float* __restrict__ dinv) {
    const int t = blockIdx.x * blockDim.x + threadIdx.x;
    if (t >= BATCH * HOUT) return;
    const int b = t / HOUT, x = t % HOUT;
    const float* hp  = H  + (size_t)b * HIN * HOUT + x;
    const float* h2p = H2 + (size_t)b * HIN * HOUT + x;
    const float tvar = wsTvar[0];
    float s = 0.f, s2 = 0.f;
    for (int d = 0; d < TDIM; ++d) { s += hp[(size_t)d * HOUT]; s2 += h2p[(size_t)d * HOUT]; }
    float* dp = dinv + (size_t)b * HOUT * HOUT + x;
    for (int y = 0; y < HOUT; ++y) {
        float var = s2 - s * s * (1.f / (float)TN) + EPSF;
        if (var < 0.f) var = 0.f;
        dp[(size_t)y * HOUT] = 1.f / sqrtf(tvar * var);
        if (y < HOUT - 1) {
            s  += hp [(size_t)(y + TDIM) * HOUT] - hp [(size_t)y * HOUT];
            s2 += h2p[(size_t)(y + TDIM) * HOUT] - h2p[(size_t)y * HOUT];
        }
    }
}

// ---------------------------------------------------------------------------
// Kernel 5: WMMA numerator + fused normalization.
// Block = 2 waves; TDM stages a 79x256-half tile (pitch 264 via TDM padding:
// interval=128 DW (one row), amount=4 DW -> 132-bank stride, conflict-free).
// Each wave owns 4 x-subtiles (4 accumulators) -> one B fragment feeds 4 WMMAs.
// ---------------------------------------------------------------------------
#define PITCH  264   // halves
#define TROWS  79
#define TCOLSD 256   // halves per staged row

__global__ void __launch_bounds__(64)
tm_wmma(const _Float16* __restrict__ Xh, const _Float16* __restrict__ wsB,
        const float* __restrict__ dinv, float* __restrict__ out) {
    __shared__ __align__(16) _Float16 tile[TROWS * PITCH];
    const int tid  = threadIdx.x;
    const int lane = tid & 31;
    const int warp = tid >> 5;
    const int b  = blockIdx.z;
    const int y0 = blockIdx.y * 16;
    const int x0 = blockIdx.x * 128;

    if (warp == 0) {
        // ---- Tensor DMA descriptor (D#), ISA 8.3/8.4 ----
        const unsigned ldsOff = (unsigned)(uintptr_t)(void*)&tile[0];
        const unsigned long long ga =
            (unsigned long long)(uintptr_t)(Xh + ((size_t)b * HIN + y0) * WIN + x0);
        u32x4 g0;
        g0.x = 1u;                                                   // count=1
        g0.y = ldsOff;                                               // lds_addr
        g0.z = (unsigned)(ga & 0xFFFFFFFFu);                         // global_addr lo
        g0.w = (unsigned)((ga >> 32) & 0x1FFFFFFu) | (2u << 30);     // addr[56:32] | type=2
        const int td0 = WIN - x0;          // remaining cols: OOB -> zero fill
        const int td1 = HIN - y0;          // remaining rows
        i32x8 g1;
        g1[0] = (1 << 16) | (1 << 20) | (6 << 22) | (3 << 25);       // 2B elems, pad en,
                                                                     // interval=128DW, amt=4DW
        g1[1] = (td0 & 0xFFFF) << 16;                                // dim0[15:0] (no atomic bar)
        g1[2] = ((td0 >> 16) & 0xFFFF) | ((td1 & 0xFFFF) << 16);     // dim0 hi | dim1 lo
        g1[3] = ((td1 >> 16) & 0xFFFF) | (TCOLSD << 16);             // dim1 hi | tile_dim0
        g1[4] = TROWS;                                               // tile_dim1 | tile_dim2=0
        g1[5] = WIN;                                                 // dim0_stride lo
        g1[6] = 0;                                                   // stride0 hi | stride1 lo
        g1[7] = 0;
        i32x4 gz4 = {0, 0, 0, 0};                                    // 2D: groups 2/3 unused
        i32x8 gz8 = {0, 0, 0, 0, 0, 0, 0, 0};                        // trailing group unused
        __builtin_amdgcn_tensor_load_to_lds(g0, g1, gz4, gz4, gz8, 0);
        __builtin_amdgcn_s_wait_tensorcnt(0);
    }
    __syncthreads();

    const int m   = lane & 15;     // A row
    const int h   = lane >> 4;     // K half-group
    const int wxw = warp * 64;     // wave's 64-column window

    v8f zero = {0.f,0.f,0.f,0.f,0.f,0.f,0.f,0.f};
    v8f acc[4] = {zero, zero, zero, zero};

    for (int dy = 0; dy < TDIM; ++dy) {
        const _Float16* arow = &tile[(m + dy) * PITCH + wxw + 16 * h];
        const _Float16* brow = wsB + (size_t)dy * (3 * 32 * 16) + lane * 16;
        #pragma unroll
        for (int c = 0; c < 3; ++c) {
            const _Float16* bp = brow + c * 512;
            v8h b0 = *(const v8h*)(bp);
            v8h b1 = *(const v8h*)(bp + 8);
            v16h B = __builtin_shufflevector(b0, b1,
                     0,1,2,3,4,5,6,7,8,9,10,11,12,13,14,15);
            #pragma unroll
            for (int t = 0; t < 4; ++t) {
                const _Float16* ap = arow + 16 * t + 32 * c;
                v8h a0 = *(const v8h*)(ap);
                v8h a1 = *(const v8h*)(ap + 8);
                v16h A = __builtin_shufflevector(a0, a1,
                         0,1,2,3,4,5,6,7,8,9,10,11,12,13,14,15);
                acc[t] = __builtin_amdgcn_wmma_f32_16x16x32_f16(
                             false, A, false, B, (short)0, acc[t], false, false);
            }
        }
    }

    const int n  = lane & 15;
    const int hc = lane >> 4;
    #pragma unroll
    for (int t = 0; t < 4; ++t) {
        const int xg = x0 + wxw + 16 * t + n;
        if (xg < HOUT) {
            #pragma unroll
            for (int v = 0; v < 8; ++v) {
                const int yg = y0 + v + 8 * hc;
                if (yg < HOUT) {
                    const size_t o = ((size_t)b * HOUT + yg) * HOUT + xg;
                    float r = acc[t][v] * dinv[o];
                    if (__builtin_isnan(r)) r = 0.f;
                    out[o] = r;
                }
            }
        }
    }
}

// ---------------------------------------------------------------------------
extern "C" void kernel_launch(void* const* d_in, const int* in_sizes, int n_in,
                              void* d_out, int out_size, void* d_ws, size_t ws_size,
                              hipStream_t stream) {
    const float* X   = (const float*)d_in[0];
    const float* tpl = (const float*)d_in[1];
    float* out = (float*)d_out;

    char* ws = (char*)d_ws;
    float*    wsTvar = (float*)(ws + OFF_TVAR);
    _Float16* wsB    = (_Float16*)(ws + OFF_B);
    float*    H      = (float*)(ws + OFF_H);
    float*    H2     = (float*)(ws + OFF_H2);
    float*    dinv   = (float*)(ws + OFF_D);
    _Float16* Xh     = (_Float16*)(ws + OFF_XH);

    tm_prep<<<1, 256, 0, stream>>>(tpl, wsTvar, wsB);
    const size_t nx4 = (size_t)BATCH * HIN * WIN / 4;       // 2,097,152
    tm_cvt<<<(unsigned)((nx4 + 255) / 256), 256, 0, stream>>>(X, Xh);
    tm_hbox<<<BATCH * HIN, 128, 0, stream>>>(X, H, H2);
    const int vthreads = BATCH * HOUT;
    tm_vbox<<<(vthreads + 255) / 256, 256, 0, stream>>>(H, H2, wsTvar, dinv);
    dim3 g(4, 29, BATCH);    // 4 x-tiles of 128 cols, 29 y-tiles of 16, 32 batches
    tm_wmma<<<g, 64, 0, stream>>>(Xh, wsB, dinv, out);
}
